// RelationMPNN_61942018343640
// MI455X (gfx1250) — matrix operand order repaired and checked by reading, not compiled, not run on hardware
//
#include <hip/hip_runtime.h>

// ---------------- problem constants (from the reference) ----------------
#define NN 20000   // nodes
#define FF 128     // node feature dim
#define HH 128     // hidden dim
#define CV 4       // C (coord channels)
#define EE 32768   // edges per relation
#define RR 8       // relations
// padded K dims (multiples of 32 for WMMA K=32)
#define KMSG 288   // 2F + C*C + ED = 273 -> 288
#define KEIN 288   // F + ED + F = 257 -> 288
#define KNOD 256   // F + H

typedef __attribute__((ext_vector_type(16))) __bf16 v16bf;
typedef __attribute__((ext_vector_type(8)))  __bf16 v8bf;
typedef __attribute__((ext_vector_type(8)))  float  v8f;
typedef __attribute__((ext_vector_type(4)))  float  v4f;

__device__ __forceinline__ float silu_f(float x) { return x / (1.f + __expf(-x)); }

// ---------------- utility kernels ----------------
__global__ void k_zero(float* p, long long n) {
  long long i = (long long)blockIdx.x * blockDim.x + threadIdx.x;
  if (i < n) p[i] = 0.f;
}

// transpose + bf16-convert (+K pad) weights: src (batch, K, Nout) -> dst (batch, Nout, Kpad)
__global__ void k_wtrans(const float* src, __bf16* dst, int K, int Nout, int Kpad, int batch) {
  long long total = (long long)batch * Nout * Kpad;
  long long i = (long long)blockIdx.x * blockDim.x + threadIdx.x;
  if (i >= total) return;
  int b = (int)(i / ((long long)Nout * Kpad));
  long long rem = i - (long long)b * Nout * Kpad;
  int n = (int)(rem / Kpad);
  int k = (int)(rem % Kpad);
  float v = (k < K) ? src[(long long)b * K * Nout + (long long)k * Nout + n] : 0.f;
  dst[i] = (__bf16)v;
}

// ---------------- geometry: cd, radial, sum over E of radial^2 ----------------
__global__ void k_geom(const float* __restrict__ coord, const int* __restrict__ rows,
                       const int* __restrict__ cols, float* __restrict__ cdb,
                       float* __restrict__ radb, float* __restrict__ sumsq, int E) {
  __shared__ float ssum[16];
  if (threadIdx.x < 16) ssum[threadIdx.x] = 0.f;
  __syncthreads();
  int e = blockIdx.x * blockDim.x + threadIdx.x;
  if (e < E) {
    int ri = rows[e], ci = cols[e];
    const float* cr = coord + (long long)ri * (CV * 3);
    const float* cc = coord + (long long)ci * (CV * 3);
    float cd[CV][3];
#pragma unroll
    for (int c = 0; c < CV; ++c)
#pragma unroll
      for (int a = 0; a < 3; ++a) {
        cd[c][a] = cr[c * 3 + a] - cc[c * 3 + a];
        cdb[(long long)e * 12 + c * 3 + a] = cd[c][a];
      }
#pragma unroll
    for (int c = 0; c < CV; ++c)
#pragma unroll
      for (int d = 0; d < CV; ++d) {
        float v = cd[c][0] * cd[d][0] + cd[c][1] * cd[d][1] + cd[c][2] * cd[d][2];
        radb[(long long)e * 16 + c * 4 + d] = v;
        atomicAdd(&ssum[c * 4 + d], v * v);   // ds_add_f32
      }
  }
  __syncthreads();
  if (threadIdx.x < 16) atomicAdd(&sumsq[threadIdx.x], ssum[threadIdx.x]);
}

// ---------------- gather msg_in (bf16, K padded to 288); one wave per edge ----------------
__global__ void k_build_msg(const float* __restrict__ h, const int* __restrict__ rows,
                            const int* __restrict__ cols, const float* __restrict__ radb,
                            const float* __restrict__ sumsq, const float* __restrict__ ea,
                            __bf16* __restrict__ msgA, int E) {
  int wave = threadIdx.x >> 5, lane = threadIdx.x & 31;
  int e = blockIdx.x * 8 + wave;
  if (e >= E) return;
  int ri = rows[e], ci = cols[e];
  __bf16* dst = msgA + (long long)e * KMSG;
  for (int c = lane; c < KMSG; c += 32) {
    float v;
    if (c < 128)       v = h[(long long)ri * FF + c];
    else if (c < 256)  v = h[(long long)ci * FF + (c - 128)];
    else if (c < 272) { int j = c - 256; v = radb[(long long)e * 16 + j] / fmaxf(sqrtf(sumsq[j]), 1e-12f); }
    else if (c == 272) v = ea[e];
    else               v = 0.f;
    dst[c] = (__bf16)v;
  }
}

// ---------------- gather e_in (bf16, K padded to 288); one wave per edge ----------------
__global__ void k_build_e(const __bf16* __restrict__ hnew, const int* __restrict__ rows,
                          const int* __restrict__ cols, const float* __restrict__ ea,
                          __bf16* __restrict__ eA, int E) {
  int wave = threadIdx.x >> 5, lane = threadIdx.x & 31;
  int e = blockIdx.x * 8 + wave;
  if (e >= E) return;
  int ri = rows[e], ci = cols[e];
  __bf16* dst = eA + (long long)e * KEIN;
  for (int c = lane; c < KEIN; c += 32) {
    float v;
    if (c < 128)       v = (float)hnew[(long long)ri * FF + c];
    else if (c == 128) v = ea[e];
    else if (c < 257)  v = (float)hnew[(long long)ci * FF + (c - 129)];
    else               v = 0.f;
    dst[c] = (__bf16)v;
  }
}

// ---------------- WMMA fragment loaders (branch-free; rows clamped to M-1) ----------------
// A 16x32 bf16 fragment (ISA: lane m=l&15, kbase=(l>>4)*8; K = kbase+[0,8) U kbase+16+[0,8))
__device__ __forceinline__ v16bf frag_a_bf16(const __bf16* __restrict__ A, long long rowOff,
                                             int kb) {
  const v8bf* p0 = (const v8bf*)(A + rowOff + kb);
  const v8bf* p1 = (const v8bf*)(A + rowOff + kb + 16);
  v8bf lo = *p0, hi = *p1;
  v16bf f;
#pragma unroll
  for (int i = 0; i < 8; ++i) { f[i] = lo[i]; f[8 + i] = hi[i]; }
  return f;
}
__device__ __forceinline__ v16bf frag_a_f32(const float* __restrict__ A, long long rowOff,
                                            int kb) {
  const v4f* p = (const v4f*)(A + rowOff + kb);
  v4f a0 = p[0], a1 = p[1];
  const v4f* q = (const v4f*)(A + rowOff + kb + 16);
  v4f a2 = q[0], a3 = q[1];
  v16bf f;
#pragma unroll
  for (int i = 0; i < 4; ++i) {
    f[i]      = (__bf16)a0[i];
    f[4 + i]  = (__bf16)a1[i];
    f[8 + i]  = (__bf16)a2[i];
    f[12 + i] = (__bf16)a3[i];
  }
  return f;
}
// B 32x16 bf16 fragment from transposed weights in LDS (lane n=l&15; K = kbase+[0,8) U kbase+16+[0,8), kbase=(l>>4)*8)
__device__ __forceinline__ v16bf frag_b_lds(const __bf16* sWl, int off) {
  const v8bf* p0 = (const v8bf*)(sWl + off);
  v8bf lo = p0[0], hi = p0[2];
  v16bf f;
#pragma unroll
  for (int i = 0; i < 8; ++i) { f[i] = lo[i]; f[8 + i] = hi[i]; }
  return f;
}

__device__ __forceinline__ void store_tile(const v8f& acc, int nIdx, int m0, int lhalf,
                                           int M, int Ntot, const float* bias,
                                           const float* residual, float* outF, __bf16* outB, int act) {
  float bv = bias ? bias[nIdx] : 0.f;
#pragma unroll
  for (int j = 0; j < 8; ++j) {
    int row = m0 + j + (lhalf << 3);
    if (row < M) {
      float v = acc[j] + bv;
      if (act) v = silu_f(v);
      long long o = (long long)row * Ntot + nIdx;
      if (residual) v += residual[o];
      if (outF) outF[o] = v;
      if (outB) outB[o] = (__bf16)v;
    }
  }
}

// ---------------- generic WMMA GEMM: out = act(A @ Wt^T + bias) [+ residual] ----------------
// A: (M, KPAD) bf16 (or f32 if AISF32), Wt: (Ntot, KPAD) bf16.
// Block = 256 thr = 8 waves; tile = 256(M) x 64(N); each wave owns 32x64 (8 accumulators).
// KPAD is compile time -> fully unrolled straight-line WMMA code, no loop SALU.
template <int KPAD, int AISF32>
__global__ __launch_bounds__(256) void k_gemm(const void* __restrict__ Av,
                                              const __bf16* __restrict__ Wt,
                                              const float* __restrict__ bias,
                                              const float* __restrict__ residual,
                                              float* outF, __bf16* outB,
                                              int M, int Ntot, int act) {
  __shared__ __bf16 sW[64 * KPAD];
  int tid = threadIdx.x;
  int n0  = blockIdx.y * 64;
  constexpr int perRow = KPAD >> 3;
  for (int c = tid; c < 64 * perRow; c += 256) { // vector b128 copies of the weight panel
    int nl = c / perRow;
    int ko = (c - nl * perRow) << 3;
    *(v8bf*)(sW + nl * KPAD + ko) = *(const v8bf*)(Wt + (long long)(n0 + nl) * KPAD + ko);
  }
  __syncthreads();

  int wave = tid >> 5, lane = tid & 31;
  int m0 = blockIdx.x * 256 + wave * 32;            // 32 M-rows per wave
  // branch-free clamped A rows (garbage rows never stored)
  int r0 = min(m0 + (lane & 15), M - 1);
  int r1 = min(m0 + 16 + (lane & 15), M - 1);
  int kbA = (lane >> 4) << 3;                       // A K-base within 32-chunk
  long long ro0 = (long long)r0 * KPAD;
  long long ro1 = (long long)r1 * KPAD;
  const __bf16* sWl = sW + (lane & 15) * KPAD + ((lane >> 4) << 3); // B lane base

  v8f acc[2][4];
#pragma unroll
  for (int i = 0; i < 2; ++i)
#pragma unroll
    for (int t = 0; t < 4; ++t) acc[i][t] = (v8f){};

#pragma unroll
  for (int k0 = 0; k0 < KPAD; k0 += 32) {
    if (k0 + 32 < KPAD) {   // keep the A stream ahead of the WMMAs (global_prefetch_b8)
      __builtin_prefetch((const char*)Av + (ro0 + k0 + 32) * (AISF32 ? 4 : 2), 0, 1);
      __builtin_prefetch((const char*)Av + (ro1 + k0 + 32) * (AISF32 ? 4 : 2), 0, 1);
    }
    v16bf a0, a1;
    if (AISF32) {
      a0 = frag_a_f32((const float*)Av, ro0, k0 + kbA);
      a1 = frag_a_f32((const float*)Av, ro1, k0 + kbA);
    } else {
      a0 = frag_a_bf16((const __bf16*)Av, ro0, k0 + kbA);
      a1 = frag_a_bf16((const __bf16*)Av, ro1, k0 + kbA);
    }
#pragma unroll
    for (int t = 0; t < 4; ++t) {
      v16bf b = frag_b_lds(sWl, t * 16 * KPAD + k0);
      acc[0][t] = __builtin_amdgcn_wmma_f32_16x16x32_bf16(false, a0, false, b, (short)0,
                                                          acc[0][t], false, false);
      acc[1][t] = __builtin_amdgcn_wmma_f32_16x16x32_bf16(false, a1, false, b, (short)0,
                                                          acc[1][t], false, false);
    }
  }

  int lcol = lane & 15, lhalf = lane >> 4;
#pragma unroll
  for (int t = 0; t < 4; ++t) {
    store_tile(acc[0][t], n0 + t * 16 + lcol, m0,      lhalf, M, Ntot, bias, residual, outF, outB, act);
    store_tile(acc[1][t], n0 + t * 16 + lcol, m0 + 16, lhalf, M, Ntot, bias, residual, outF, outB, act);
  }
}

// ---------------- per-relation scatter of ef into seg (one wave / edge) ----------------
__global__ void k_segacc(const __bf16* __restrict__ efb, const int* __restrict__ rows,
                         float* __restrict__ seg, int E) {
  int wave = threadIdx.x >> 5, lane = threadIdx.x & 31;
  int e = blockIdx.x * 8 + wave;
  if (e >= E) return;
  int ri = rows[e];
  const __bf16* src = efb + (long long)e * HH;
  float* dst = seg + (long long)ri * HH;
  for (int j = lane; j < HH; j += 32) atomicAdd(&dst[j], (float)src[j]);
}

// ---------------- phi = hid2 @ Wc2[r]  (128 -> 4, VALU dot) ----------------
__global__ void k_phi(const __bf16* __restrict__ hid2, const float* __restrict__ Wc2r,
                      float* __restrict__ phib, int E) {
  int idx = blockIdx.x * blockDim.x + threadIdx.x;
  if (idx >= E * CV) return;
  int e = idx >> 2, c = idx & 3;
  const __bf16* hr = hid2 + (long long)e * HH;
  float s = 0.f;
#pragma unroll 8
  for (int j = 0; j < HH; ++j) s += (float)hr[j] * Wc2r[j * CV + c];
  phib[idx] = s;
}

// ---------------- trans = cd*phi, scatter into s/cnt ----------------
__global__ void k_trans(const float* __restrict__ cdb, const float* __restrict__ phib,
                        const int* __restrict__ rows, float* __restrict__ sacc,
                        float* __restrict__ cnt, int E) {
  int e = blockIdx.x * blockDim.x + threadIdx.x;
  if (e >= E) return;
  int ri = rows[e];
#pragma unroll
  for (int c = 0; c < CV; ++c) {
    float p = phib[(long long)e * CV + c];
#pragma unroll
    for (int a = 0; a < 3; ++a)
      atomicAdd(&sacc[(long long)ri * 12 + c * 3 + a], cdb[(long long)e * 12 + c * 3 + a] * p);
  }
  atomicAdd(&cnt[ri], 1.f);
}

// ---------------- x = coord + s / max(cnt,1) ----------------
__global__ void k_xfinal(const float* __restrict__ coord, const float* __restrict__ sacc,
                         const float* __restrict__ cnt, float* __restrict__ outx, int N) {
  int idx = blockIdx.x * blockDim.x + threadIdx.x;
  if (idx >= N * 12) return;
  int n = idx / 12;
  outx[idx] = coord[idx] + sacc[idx] / fmaxf(cnt[n], 1.f);
}

// ---------------- nodeA = concat(h, agg) as bf16 ----------------
__global__ void k_nodeA(const float* __restrict__ h, const float* __restrict__ agg,
                        __bf16* __restrict__ nodeA, int N) {
  long long idx = (long long)blockIdx.x * blockDim.x + threadIdx.x;
  if (idx >= (long long)N * KNOD) return;
  int n = (int)(idx >> 8), c = (int)(idx & 255);
  float v = (c < FF) ? h[(long long)n * FF + c] : agg[(long long)n * HH + (c - FF)];
  nodeA[idx] = (__bf16)v;
}

// ---------------- m = ehid . We2 + be2 ----------------
__global__ void k_m(const __bf16* __restrict__ ehid, const float* __restrict__ We2,
                    const float* __restrict__ be2, float* __restrict__ outm, int E) {
  int e = blockIdx.x * blockDim.x + threadIdx.x;
  if (e >= E) return;
  const __bf16* hr = ehid + (long long)e * HH;
  float s = 0.f;
#pragma unroll 8
  for (int j = 0; j < HH; ++j) s += (float)hr[j] * We2[j];
  outm[e] = s + be2[0];
}

// ==================================================================
extern "C" void kernel_launch(void* const* d_in, const int* in_sizes, int n_in,
                              void* d_out, int out_size, void* d_ws, size_t ws_size,
                              hipStream_t stream) {
  (void)in_sizes; (void)n_in; (void)out_size; (void)ws_size;
  const float* h     = (const float*)d_in[0];
  const float* coord = (const float*)d_in[1];
  const float* eattr = (const float*)d_in[2];
  const int*   edges = (const int*)d_in[3];
  const float* Wm1 = (const float*)d_in[4];  const float* bm1 = (const float*)d_in[5];
  const float* Wm2 = (const float*)d_in[6];  const float* bm2 = (const float*)d_in[7];
  const float* We1 = (const float*)d_in[8];  const float* be1 = (const float*)d_in[9];
  const float* We2 = (const float*)d_in[10]; const float* be2 = (const float*)d_in[11];
  const float* Wn1 = (const float*)d_in[12]; const float* bn1 = (const float*)d_in[13];
  const float* Wn2 = (const float*)d_in[14]; const float* bn2 = (const float*)d_in[15];
  const float* Wrel= (const float*)d_in[16];
  const float* Wc1 = (const float*)d_in[17]; const float* bc1 = (const float*)d_in[18];
  const float* Wc2 = (const float*)d_in[19];

  float* out_h = (float*)d_out;
  float* out_x = out_h + (size_t)NN * FF;
  float* out_m = out_x + (size_t)NN * CV * 3;

  // ---- workspace bump allocation (256B aligned) ----
  char* wsb = (char*)d_ws;
  size_t off = 0;
  auto alloc = [&](size_t bytes) -> char* {
    char* p = wsb + off;
    off = (off + bytes + 255) & ~(size_t)255;
    return p;
  };
  __bf16* Wm1t  = (__bf16*)alloc((size_t)HH * KMSG * 2);
  __bf16* Wm2t  = (__bf16*)alloc((size_t)HH * HH * 2);
  __bf16* Wc1t  = (__bf16*)alloc((size_t)RR * HH * HH * 2);
  __bf16* Wn1t  = (__bf16*)alloc((size_t)HH * KNOD * 2);
  __bf16* Wn2t  = (__bf16*)alloc((size_t)FF * HH * 2);
  __bf16* We1t  = (__bf16*)alloc((size_t)HH * KEIN * 2);
  __bf16* Wrelt = (__bf16*)alloc((size_t)RR * FF * HH * 2);
  __bf16* msgA  = (__bf16*)alloc((size_t)EE * KMSG * 2);   // reused for e_in
  __bf16* hid1  = (__bf16*)alloc((size_t)EE * HH * 2);     // reused (hid2, ehid)
  __bf16* efb   = (__bf16*)alloc((size_t)EE * HH * 2);
  __bf16* nodeA = (__bf16*)alloc((size_t)NN * KNOD * 2);
  __bf16* nhid  = (__bf16*)alloc((size_t)NN * HH * 2);
  __bf16* hnewb = (__bf16*)alloc((size_t)NN * FF * 2);
  float*  cdb   = (float*)alloc((size_t)EE * 12 * 4);
  float*  radb  = (float*)alloc((size_t)EE * 16 * 4);
  float*  sumsq = (float*)alloc(16 * 4);
  float*  phib  = (float*)alloc((size_t)EE * CV * 4);
  float*  seg   = (float*)alloc((size_t)NN * HH * 4);
  float*  sacc  = (float*)alloc((size_t)NN * 12 * 4);
  float*  cnt   = (float*)alloc((size_t)NN * 4);
  float*  agg   = (float*)alloc((size_t)NN * FF * 4);

  const int T = 256;
  #define GR(n) dim3((unsigned)(((long long)(n) + T - 1) / T))

  // ---- weight prep (bf16 transposed, K padded) ----
  k_wtrans<<<GR(HH * KMSG), T, 0, stream>>>(Wm1, Wm1t, 273, HH, KMSG, 1);
  k_wtrans<<<GR(HH * HH), T, 0, stream>>>(Wm2, Wm2t, HH, HH, HH, 1);
  k_wtrans<<<GR(RR * HH * HH), T, 0, stream>>>(Wc1, Wc1t, HH, HH, HH, RR);
  k_wtrans<<<GR(HH * KNOD), T, 0, stream>>>(Wn1, Wn1t, KNOD, HH, KNOD, 1);
  k_wtrans<<<GR(FF * HH), T, 0, stream>>>(Wn2, Wn2t, HH, FF, HH, 1);
  k_wtrans<<<GR(HH * KEIN), T, 0, stream>>>(We1, We1t, 257, HH, KEIN, 1);
  k_wtrans<<<GR(RR * FF * HH), T, 0, stream>>>(Wrel, Wrelt, HH, FF, HH, RR);

  // ---- zero atomic accumulators ----
  k_zero<<<GR(NN * 12), T, 0, stream>>>(sacc, (long long)NN * 12);
  k_zero<<<GR(NN), T, 0, stream>>>(cnt, NN);
  k_zero<<<GR(NN * FF), T, 0, stream>>>(agg, (long long)NN * FF);

  dim3 ge((EE + 255) / 256, 2);   // edge GEMMs: 256-row blocks, Ntot=128 -> 2 col blocks
  dim3 gn((NN + 255) / 256, 2);   // node GEMMs

  // ---- per-relation message/coordinate phase ----
  for (int r = 0; r < RR; ++r) {
    const int* rows = edges + (size_t)r * 2 * EE;
    const int* cols = rows + EE;
    const float* ear = eattr + (size_t)r * EE;
    k_zero<<<1, T, 0, stream>>>(sumsq, 16);
    k_zero<<<GR(NN * HH), T, 0, stream>>>(seg, (long long)NN * HH);
    k_geom<<<GR(EE), T, 0, stream>>>(coord, rows, cols, cdb, radb, sumsq, EE);
    k_build_msg<<<EE / 8, T, 0, stream>>>(h, rows, cols, radb, sumsq, ear, msgA, EE);
    k_gemm<KMSG, 0><<<ge, T, 0, stream>>>(msgA, Wm1t, bm1, nullptr, nullptr, hid1, EE, HH, 1);
    k_gemm<HH, 0><<<ge, T, 0, stream>>>(hid1, Wm2t, bm2, nullptr, nullptr, efb, EE, HH, 1);
    k_segacc<<<EE / 8, T, 0, stream>>>(efb, rows, seg, EE);
    k_gemm<HH, 0><<<ge, T, 0, stream>>>(efb, Wc1t + (size_t)r * HH * HH, bc1 + (size_t)r * HH,
                                        nullptr, nullptr, hid1, EE, HH, 1);
    k_phi<<<GR(EE * CV), T, 0, stream>>>(hid1, Wc2 + (size_t)r * HH * CV, phib, EE);
    k_trans<<<GR(EE), T, 0, stream>>>(cdb, phib, rows, sacc, cnt, EE);
    // agg += seg @ Wrel[r]   (A is f32, converted to bf16 in the fragment loader)
    k_gemm<HH, 1><<<gn, T, 0, stream>>>(seg, Wrelt + (size_t)r * FF * HH, nullptr,
                                        agg, agg, nullptr, NN, FF, 0);
  }

  // ---- coordinates out ----
  k_xfinal<<<GR(NN * 12), T, 0, stream>>>(coord, sacc, cnt, out_x, NN);

  // ---- node MLP: h_new = h + MLP(concat(h, agg)) ----
  k_nodeA<<<GR((long long)NN * KNOD), T, 0, stream>>>(h, agg, nodeA, NN);
  k_gemm<KNOD, 0><<<gn, T, 0, stream>>>(nodeA, Wn1t, bn1, nullptr, nullptr, nhid, NN, HH, 1);
  k_gemm<HH, 0><<<gn, T, 0, stream>>>(nhid, Wn2t, bn2, h, out_h, hnewb, NN, FF, 0);

  // ---- per-relation edge-output phase ----
  for (int r = 0; r < RR; ++r) {
    const int* rows = edges + (size_t)r * 2 * EE;
    const int* cols = rows + EE;
    const float* ear = eattr + (size_t)r * EE;
    k_build_e<<<EE / 8, T, 0, stream>>>(hnewb, rows, cols, ear, msgA, EE);
    k_gemm<KEIN, 0><<<ge, T, 0, stream>>>(msgA, We1t, be1, nullptr, nullptr, hid1, EE, HH, 1);
    k_m<<<GR(EE), T, 0, stream>>>(hid1, We2, be2, out_m + (size_t)r * EE, EE);
  }
  #undef GR
}
